// RelationTransformerEncoderBlock_26860725469641
// MI455X (gfx1250) — matrix operand order repaired and checked by reading, not compiled, not run on hardware
//
#include <hip/hip_runtime.h>
#include <cstdint>
#include <cstddef>

#define USE_ASYNC_STAGING 1

// ---------------- problem constants ----------------
#define BQ   2
#define TQ   1024
#define DQ   1024
#define HQ   16
#define HDQ  64
#define AQ   1024            // H*HD
#define FQ   4096
#define BTQ  (BQ*TQ)         // 2048
#define BHQ  (BQ*HQ)         // 32
#define LN_EPS 1e-5f
#define MASK_FILL 1e-32f

// ---------------- vector types ----------------
typedef __attribute__((ext_vector_type(16))) __bf16 v16bf;
typedef __attribute__((ext_vector_type(8)))  __bf16 v8bf;
typedef __attribute__((ext_vector_type(8)))  float  v8f;

static __device__ __forceinline__ __bf16 f2bf(float f) {
  union { float f; unsigned u; } in; in.f = f;
  unsigned u = in.u;
  unsigned short b = (unsigned short)((u + 0x7FFFu + ((u >> 16) & 1u)) >> 16);
  union { unsigned short s; __bf16 h; } out; out.s = b;
  return out.h;
}

#if USE_ASYNC_STAGING
#define ASYNC_B128(ldsoff, gaddr)                                          \
  asm volatile("global_load_async_to_lds_b128 %0, %1, off"                 \
               :: "v"(ldsoff), "v"(gaddr) : "memory")
#define ASYNC_WAIT() asm volatile("s_wait_asynccnt 0x0" ::: "memory")
#else
#define ASYNC_WAIT()
#endif

// =====================================================================
// Batched bf16 WMMA GEMM:  C[z] = alpha * A[z] @ Bt[z]^T  (+bias)
//   A : [M,K] bf16 row-major     (M % 256 == 0, K % 32 == 0)
//   Bt: [N,K] bf16 row-major  -> logical B = Bt^T
//   C : f32 or bf16; per-batch base = (z>>cShift)*sCb + (z&(2^cShift-1))*sCh
//   MCOL/MROW: int mask (0 -> maskFill) indexed by column / row.
//   Block tile 256x128, 256 threads = 8 waves (4x2), wave tile 64x64:
//   16 WMMAs per 16 ds_load_b128 per K-step. Double-buffered LDS fed by
//   global_load_async_to_lds_b128 (ASYNCcnt), global_prefetch 2 steps out.
// =====================================================================
#define BM 256
#define BN 128
#define BK 32
#define SA 40   // LDS row stride (BK + 8 pad) -> 80B rows, 16B aligned chunks
#define LDSBUF_A (BM * SA)
#define LDSBUF_B (BN * SA)

template<bool OUT_BF16, bool RELU, bool MCOL, bool MROW>
__global__ __launch_bounds__(256)
void gemm_wmma_bf16(const __bf16* __restrict__ A, const __bf16* __restrict__ Bt,
                    void* __restrict__ C,
                    long long sA, long long sB, long long sCb, long long sCh,
                    int cShift, int lda, int ldb, int ldc, int N, int K,
                    const float* __restrict__ bias, float alpha,
                    const int* __restrict__ mask, int mShift,
                    long long maskStride, float maskFill)
{
  __shared__ __attribute__((aligned(16))) __bf16 As[2][LDSBUF_A];
  __shared__ __attribute__((aligned(16))) __bf16 Bs[2][LDSBUF_B];

  const int tid  = threadIdx.x;
  const int lane = tid & 31;
  const int wave = tid >> 5;
  const int wm   = wave >> 1;     // 0..3 (64-row slab)
  const int wn   = wave & 1;      // 0..1 (64-col slab)
  const int lr   = lane & 15;
  const int hsel = lane >> 4;

  const int z  = blockIdx.z;
  const int m0 = blockIdx.y * BM;
  const int n0 = blockIdx.x * BN;

  const __bf16* Ap = A  + (size_t)z * (size_t)sA + (size_t)m0 * lda;
  const __bf16* Bp = Bt + (size_t)z * (size_t)sB + (size_t)n0 * ldb;
  const int* mptr = nullptr;
  if (MCOL || MROW)
    mptr = mask + (size_t)(z >> mShift) * (size_t)maskStride;

  // staging decomposition: thread covers 8 bf16 at col kc0, rows r0+i*64
  const int r0  = tid >> 2;
  const int kc0 = (tid & 3) * 8;
#if USE_ASYNC_STAGING
  const unsigned asLds = (unsigned)(size_t)&As[0][0];
  const unsigned bsLds = (unsigned)(size_t)&Bs[0][0];
#endif

  v8f acc[4][4];
  v8f zf = {};
#pragma unroll
  for (int mi = 0; mi < 4; ++mi)
#pragma unroll
    for (int ni = 0; ni < 4; ++ni) acc[mi][ni] = zf;

  auto stage = [&](int kt, int buf) {
#pragma unroll
    for (int i = 0; i < 4; ++i) {          // A: 256 rows
      int r = r0 + i * 64;
      const __bf16* ga = Ap + (size_t)r * lda + kt * BK + kc0;
#if USE_ASYNC_STAGING
      unsigned la = asLds + (unsigned)((buf * LDSBUF_A + r * SA + kc0) * 2);
      ASYNC_B128(la, (unsigned long long)(size_t)ga);
#else
      *(v8bf*)&As[buf][r * SA + kc0] = *(const v8bf*)ga;
#endif
    }
#pragma unroll
    for (int i = 0; i < 2; ++i) {          // B: 128 rows
      int r = r0 + i * 64;
      if (n0 + r < N) {
        const __bf16* gb = Bp + (size_t)r * ldb + kt * BK + kc0;
#if USE_ASYNC_STAGING
        unsigned lb = bsLds + (unsigned)((buf * LDSBUF_B + r * SA + kc0) * 2);
        ASYNC_B128(lb, (unsigned long long)(size_t)gb);
#else
        *(v8bf*)&Bs[buf][r * SA + kc0] = *(const v8bf*)gb;
#endif
      }
    }
  };

  auto compute = [&](int buf) {
    const __bf16* Ab = &As[buf][0];
    const __bf16* Bb = &Bs[buf][0];
    v16bf afrag[4], bfrag[4];
#pragma unroll
    for (int mi = 0; mi < 4; ++mi) {
      const __bf16* p = Ab + (wm * 64 + mi * 16 + lr) * SA + hsel * 8;
      v8bf lo = *(const v8bf*)p;
      v8bf hi = *(const v8bf*)(p + 16);
      afrag[mi] = __builtin_shufflevector(lo, hi,
          0,1,2,3,4,5,6,7,8,9,10,11,12,13,14,15);
    }
#pragma unroll
    for (int ni = 0; ni < 4; ++ni) {
      const __bf16* p = Bb + (wn * 64 + ni * 16 + lr) * SA + hsel * 8;
      v8bf lo = *(const v8bf*)p;
      v8bf hi = *(const v8bf*)(p + 16);
      bfrag[ni] = __builtin_shufflevector(lo, hi,
          0,1,2,3,4,5,6,7,8,9,10,11,12,13,14,15);
    }
#pragma unroll
    for (int mi = 0; mi < 4; ++mi)
#pragma unroll
      for (int ni = 0; ni < 4; ++ni)
        acc[mi][ni] = __builtin_amdgcn_wmma_f32_16x16x32_bf16(
            false, afrag[mi], false, bfrag[ni], (short)0, acc[mi][ni],
            false, false);
  };

  const int nk = K / BK;
  stage(0, 0);
  ASYNC_WAIT();
  __syncthreads();
  for (int kt = 0; kt < nk; ++kt) {
    const int cur = kt & 1;
    if (kt + 1 < nk) stage(kt + 1, cur ^ 1);
    if (kt + 2 < nk) {   // warm L2 for the async engine
      __builtin_prefetch(Ap + (size_t)r0 * lda + (kt + 2) * BK + kc0, 0, 1);
      if (n0 + r0 < N)
        __builtin_prefetch(Bp + (size_t)r0 * ldb + (kt + 2) * BK + kc0, 0, 1);
    }
    compute(cur);
    if (kt + 1 < nk) {
      ASYNC_WAIT();
      __syncthreads();
    }
  }

  // ---- epilogue ----
  const int zHi = z >> cShift;
  const int zLo = z & ((1 << cShift) - 1);
  size_t cbase = (size_t)zHi * (size_t)sCb + (size_t)zLo * (size_t)sCh;
#pragma unroll
  for (int mi = 0; mi < 4; ++mi) {
#pragma unroll
    for (int ni = 0; ni < 4; ++ni) {
      int colb = n0 + wn * 64 + ni * 16 + lr;
      if (colb < N) {
#pragma unroll
        for (int r = 0; r < 8; ++r) {
          int row = m0 + wm * 64 + mi * 16 + hsel * 8 + r;
          float v = acc[mi][ni][r] * alpha;
          if (bias) v += bias[colb];
          if (MCOL) { if (mptr[colb] == 0) v = maskFill; }
          if (MROW) { if (mptr[row]  == 0) v = maskFill; }
          if (RELU) v = fmaxf(v, 0.0f);
          if (OUT_BF16)
            ((__bf16*)C)[cbase + (size_t)row * ldc + colb] = f2bf(v);
          else
            ((float*)C)[cbase + (size_t)row * ldc + colb] = v;
        }
      }
    }
  }
}

// =====================================================================
// Wave32 reductions
// =====================================================================
static __device__ __forceinline__ float wave_max(float v) {
#pragma unroll
  for (int m = 16; m > 0; m >>= 1) v = fmaxf(v, __shfl_xor(v, m, 32));
  return v;
}
static __device__ __forceinline__ float wave_sum(float v) {
#pragma unroll
  for (int m = 16; m > 0; m >>= 1) v += __shfl_xor(v, m, 32);
  return v;
}

// =====================================================================
// Elementwise / reduction kernels
// =====================================================================
__global__ void f32_to_bf16_kernel(const float* __restrict__ x,
                                   __bf16* __restrict__ y, long long n) {
  long long i = (long long)blockIdx.x * blockDim.x + threadIdx.x;
  if (i < n) y[i] = f2bf(x[i]);
}

// x: [R,C] f32 -> y: [C,R] bf16 (tiled transpose)
__global__ __launch_bounds__(256)
void f32_to_bf16_T_kernel(const float* __restrict__ x, __bf16* __restrict__ y,
                          int R, int Cc) {
  __shared__ float tile[32][33];
  int r0 = blockIdx.y * 32, c0 = blockIdx.x * 32;
  int tx = threadIdx.x & 31, ty = threadIdx.x >> 5;
#pragma unroll
  for (int i = ty; i < 32; i += 8)
    tile[i][tx] = x[(size_t)(r0 + i) * Cc + c0 + tx];
  __syncthreads();
#pragma unroll
  for (int i = ty; i < 32; i += 8)
    y[(size_t)(c0 + i) * R + r0 + tx] = f2bf(tile[tx][i]);
}

// qkv f32 [B,T,3A] -> q,k bf16 [B,H,T,HD], vT bf16 [B,H,HD,T]
__global__ void split_qkv_kernel(const float* __restrict__ qkv,
                                 __bf16* __restrict__ q,
                                 __bf16* __restrict__ k,
                                 __bf16* __restrict__ vT) {
  long long idx = (long long)blockIdx.x * blockDim.x + threadIdx.x;
  if (idx >= (long long)BTQ * AQ) return;
  int b = (int)(idx / ((long long)TQ * AQ));
  int r = (int)(idx % ((long long)TQ * AQ));
  int t = r / AQ;
  int a = r % AQ;
  int h = a / HDQ;
  int d = a % HDQ;
  const float* src = qkv + ((size_t)b * TQ + t) * (3 * AQ);
  size_t zh = (size_t)b * HQ + h;
  q [(zh * TQ + t) * HDQ + d] = f2bf(src[a]);
  k [(zh * TQ + t) * HDQ + d] = f2bf(src[AQ + a]);
  vT[(zh * HDQ + d) * TQ + t] = f2bf(src[2 * AQ + a]);
}

// row softmax: X f32 [rows, cols] -> Y bf16
__global__ __launch_bounds__(256)
void softmax_rows_kernel(const float* __restrict__ X, __bf16* __restrict__ Y,
                         int cols) {
  __shared__ float red[8];
  const int tid = threadIdx.x, lane = tid & 31, wave = tid >> 5;
  const float* x = X + (size_t)blockIdx.x * cols;
  __bf16* y = Y + (size_t)blockIdx.x * cols;

  float m = -3.4e38f;
  for (int c = tid; c < cols; c += 256) m = fmaxf(m, x[c]);
  m = wave_max(m);
  if (lane == 0) red[wave] = m;
  __syncthreads();
  m = red[0];
#pragma unroll
  for (int j = 1; j < 8; ++j) m = fmaxf(m, red[j]);
  __syncthreads();

  float sum = 0.0f;
  for (int c = tid; c < cols; c += 256) sum += __expf(x[c] - m);
  sum = wave_sum(sum);
  if (lane == 0) red[wave] = sum;
  __syncthreads();
  sum = red[0];
#pragma unroll
  for (int j = 1; j < 8; ++j) sum += red[j];
  float inv = 1.0f / sum;
  for (int c = tid; c < cols; c += 256) y[c] = f2bf(__expf(x[c] - m) * inv);
}

// out = LayerNorm(X + R) * g + be ; optional bf16 mirror
__global__ __launch_bounds__(256)
void add_ln_kernel(const float* __restrict__ X, const float* __restrict__ R,
                   const float* __restrict__ g, const float* __restrict__ be,
                   float* __restrict__ outF, __bf16* __restrict__ outB,
                   int cols) {
  __shared__ float redS[8];
  __shared__ float redQ[8];
  const int tid = threadIdx.x, lane = tid & 31, wave = tid >> 5;
  const float* x = X + (size_t)blockIdx.x * cols;
  const float* r = R + (size_t)blockIdx.x * cols;

  float s = 0.0f, q = 0.0f;
  for (int c = tid; c < cols; c += 256) {
    float v = x[c] + r[c];
    s += v; q += v * v;
  }
  s = wave_sum(s); q = wave_sum(q);
  if (lane == 0) { redS[wave] = s; redQ[wave] = q; }
  __syncthreads();
  s = redS[0]; q = redQ[0];
#pragma unroll
  for (int j = 1; j < 8; ++j) { s += redS[j]; q += redQ[j]; }

  float mean = s / cols;
  float var  = q / cols - mean * mean;
  float rstd = rsqrtf(var + LN_EPS);

  float* of = outF + (size_t)blockIdx.x * cols;
  __bf16* ob = outB ? outB + (size_t)blockIdx.x * cols : nullptr;
  for (int c = tid; c < cols; c += 256) {
    float v = (x[c] + r[c] - mean) * rstd * g[c] + be[c];
    of[c] = v;
    if (ob) ob[c] = f2bf(v);
  }
}

// =====================================================================
// Host-side orchestration
// =====================================================================
enum GemmMode { GM_F32 = 0, GM_BF16, GM_BF16_RELU, GM_BF16_MCOL, GM_BF16_MROW };

static inline void launch_gemm(hipStream_t st,
                               const __bf16* A, const __bf16* Bt, void* C,
                               long long sA, long long sB,
                               long long sCb, long long sCh, int cShift,
                               int lda, int ldb, int ldc,
                               int M, int N, int K, int batch,
                               const float* bias, float alpha,
                               const int* mask, int mShift, long long maskStride,
                               float maskFill, GemmMode mode) {
  dim3 grid((N + BN - 1) / BN, (M + BM - 1) / BM, batch);
  dim3 block(256);
  switch (mode) {
    case GM_F32:
      gemm_wmma_bf16<false, false, false, false><<<grid, block, 0, st>>>(
          A, Bt, C, sA, sB, sCb, sCh, cShift, lda, ldb, ldc, N, K,
          bias, alpha, nullptr, 0, 0, 0.0f);
      break;
    case GM_BF16:
      gemm_wmma_bf16<true, false, false, false><<<grid, block, 0, st>>>(
          A, Bt, C, sA, sB, sCb, sCh, cShift, lda, ldb, ldc, N, K,
          bias, alpha, nullptr, 0, 0, 0.0f);
      break;
    case GM_BF16_RELU:
      gemm_wmma_bf16<true, true, false, false><<<grid, block, 0, st>>>(
          A, Bt, C, sA, sB, sCb, sCh, cShift, lda, ldb, ldc, N, K,
          bias, alpha, nullptr, 0, 0, 0.0f);
      break;
    case GM_BF16_MCOL:
      gemm_wmma_bf16<true, false, true, false><<<grid, block, 0, st>>>(
          A, Bt, C, sA, sB, sCb, sCh, cShift, lda, ldb, ldc, N, K,
          bias, alpha, mask, mShift, maskStride, maskFill);
      break;
    case GM_BF16_MROW:
      gemm_wmma_bf16<true, false, false, true><<<grid, block, 0, st>>>(
          A, Bt, C, sA, sB, sCb, sCh, cShift, lda, ldb, ldc, N, K,
          bias, alpha, mask, mShift, maskStride, maskFill);
      break;
  }
}

extern "C" void kernel_launch(void* const* d_in, const int* in_sizes, int n_in,
                              void* d_out, int out_size, void* d_ws, size_t ws_size,
                              hipStream_t stream) {
  (void)in_sizes; (void)n_in; (void)out_size; (void)ws_size;

  const float* src = (const float*)d_in[0];
  const float* aux = (const float*)d_in[1];
  const float* Wu  = (const float*)d_in[2];
  const float* bu  = (const float*)d_in[3];
  const float* Wv  = (const float*)d_in[4];
  const float* bv  = (const float*)d_in[5];
  const float* Wo  = (const float*)d_in[6];
  const float* bo  = (const float*)d_in[7];
  const float* W1  = (const float*)d_in[8];
  const float* b1  = (const float*)d_in[9];
  const float* W2  = (const float*)d_in[10];
  const float* b2  = (const float*)d_in[11];
  const float* g1  = (const float*)d_in[12];
  const float* be1 = (const float*)d_in[13];
  const float* g2  = (const float*)d_in[14];
  const float* be2 = (const float*)d_in[15];
  const int* src_mask = (const int*)d_in[16];
  const int* aux_mask = (const int*)d_in[17];
  float* outp = (float*)d_out;

  // ---- workspace bump allocator ----
  char* ws = (char*)d_ws;
  size_t off = 0;
  auto alloc = [&](size_t bytes) -> char* {
    char* p = ws + off;
    off = (off + bytes + 255) & ~(size_t)255;
    return p;
  };
  const size_t TT = (size_t)TQ * TQ;

  __bf16* src_bf = (__bf16*)alloc((size_t)BTQ * DQ * 2);
  __bf16* aux_bf = (__bf16*)alloc((size_t)BTQ * DQ * 2);
  __bf16* WuT    = (__bf16*)alloc((size_t)DQ * 3 * AQ * 2);  // [3A, D]
  __bf16* WvT    = (__bf16*)alloc((size_t)DQ * 3 * AQ * 2);  // [3A, D]
  __bf16* WoT    = (__bf16*)alloc((size_t)AQ * DQ * 2);      // [D, A]
  __bf16* W1T    = (__bf16*)alloc((size_t)DQ * FQ * 2);      // [F, D]
  __bf16* W2T    = (__bf16*)alloc((size_t)FQ * DQ * 2);      // [D, F]
  __bf16* q_u    = (__bf16*)alloc((size_t)BHQ * TQ * HDQ * 2);
  __bf16* k_u    = (__bf16*)alloc((size_t)BHQ * TQ * HDQ * 2);
  __bf16* vT_u   = (__bf16*)alloc((size_t)BHQ * TQ * HDQ * 2);
  __bf16* q_v    = (__bf16*)alloc((size_t)BHQ * TQ * HDQ * 2);
  __bf16* k_v    = (__bf16*)alloc((size_t)BHQ * TQ * HDQ * 2);
  __bf16* vT_v   = (__bf16*)alloc((size_t)BHQ * TQ * HDQ * 2);
  __bf16* sim1   = (__bf16*)alloc((size_t)BHQ * TT * 2);
  __bf16* sim1T  = (__bf16*)alloc((size_t)BHQ * TT * 2);
  __bf16* sim2   = (__bf16*)alloc((size_t)BHQ * TT * 2);
  __bf16* sim2T  = (__bf16*)alloc((size_t)BHQ * TT * 2);
  __bf16* att    = (__bf16*)alloc((size_t)BHQ * TT * 2);
  __bf16* attm   = (__bf16*)alloc((size_t)BTQ * AQ * 2);
  __bf16* o1b    = (__bf16*)alloc((size_t)BTQ * DQ * 2);
  __bf16* hb     = (__bf16*)alloc((size_t)BTQ * FQ * 2);
  float* qkv_u   = (float*)alloc((size_t)BTQ * 3 * AQ * 4);
  float* qkv_v   = (float*)alloc((size_t)BTQ * 3 * AQ * 4);
  float* comb    = (float*)alloc((size_t)BHQ * TT * 4);
  float* attno   = (float*)alloc((size_t)BTQ * DQ * 4);
  float* o1f     = (float*)alloc((size_t)BTQ * DQ * 4);
  float* ffn     = (float*)alloc((size_t)BTQ * DQ * 4);

  auto conv = [&](const float* x, __bf16* y, long long n) {
    f32_to_bf16_kernel<<<dim3((unsigned)((n + 255) / 256)), dim3(256), 0, stream>>>(x, y, n);
  };
  auto convT = [&](const float* x, __bf16* y, int R, int Cc) {
    f32_to_bf16_T_kernel<<<dim3(Cc / 32, R / 32), dim3(256), 0, stream>>>(x, y, R, Cc);
  };

  // ---- 1. staging: activations bf16, weights transposed bf16 ----
  conv(src, src_bf, (long long)BTQ * DQ);
  conv(aux, aux_bf, (long long)BTQ * DQ);
  convT(Wu, WuT, DQ, 3 * AQ);
  convT(Wv, WvT, DQ, 3 * AQ);
  convT(Wo, WoT, AQ, DQ);
  convT(W1, W1T, DQ, FQ);
  convT(W2, W2T, FQ, DQ);

  // ---- 2. QKV projections: qkv = x @ W + b  (Bt = W^T) ----
  launch_gemm(stream, src_bf, WuT, qkv_u, 0, 0, 0, 0, 0,
              DQ, DQ, 3 * AQ, BTQ, 3 * AQ, DQ, 1,
              bu, 1.0f, nullptr, 0, 0, 0.0f, GM_F32);
  launch_gemm(stream, aux_bf, WvT, qkv_v, 0, 0, 0, 0, 0,
              DQ, DQ, 3 * AQ, BTQ, 3 * AQ, DQ, 1,
              bv, 1.0f, nullptr, 0, 0, 0.0f, GM_F32);

  // ---- 3. split heads ----
  {
    long long n = (long long)BTQ * AQ;
    dim3 g((unsigned)((n + 255) / 256)), b(256);
    split_qkv_kernel<<<g, b, 0, stream>>>(qkv_u, q_u, k_u, vT_u);
    split_qkv_kernel<<<g, b, 0, stream>>>(qkv_v, q_v, k_v, vT_v);
  }

  // ---- 4. masked scaled similarities (+ transposed copies) ----
  const float inv_scale = 0.125f;  // 1/sqrt(64)
  const long long sQK = (long long)TQ * HDQ;
  launch_gemm(stream, q_u, k_v, sim1, sQK, sQK, (long long)TT, 0, 0,
              HDQ, HDQ, TQ, TQ, TQ, HDQ, BHQ,
              nullptr, inv_scale, aux_mask, 4, TQ, MASK_FILL, GM_BF16_MCOL);
  launch_gemm(stream, k_v, q_u, sim1T, sQK, sQK, (long long)TT, 0, 0,
              HDQ, HDQ, TQ, TQ, TQ, HDQ, BHQ,
              nullptr, inv_scale, aux_mask, 4, TQ, MASK_FILL, GM_BF16_MROW);
  launch_gemm(stream, q_v, k_u, sim2, sQK, sQK, (long long)TT, 0, 0,
              HDQ, HDQ, TQ, TQ, TQ, HDQ, BHQ,
              nullptr, inv_scale, src_mask, 4, TQ, MASK_FILL, GM_BF16_MCOL);
  launch_gemm(stream, k_u, q_v, sim2T, sQK, sQK, (long long)TT, 0, 0,
              HDQ, HDQ, TQ, TQ, TQ, HDQ, BHQ,
              nullptr, inv_scale, src_mask, 4, TQ, MASK_FILL, GM_BF16_MROW);

  // ---- 5. per-stream attention + FFN ----
  for (int sidx = 0; sidx < 2; ++sidx) {
    const __bf16* simA  = (sidx == 0) ? sim1  : sim2;   // comb = simA @ simB
    const __bf16* simBT = (sidx == 0) ? sim2T : sim1T;  // (B^T form)
    const __bf16* vT    = (sidx == 0) ? vT_u : vT_v;
    const float*  x_f   = (sidx == 0) ? src : aux;
    float* outs = outp + (size_t)sidx * BTQ * DQ;

    // comb = simA @ simB  (dominant GEMM, K = T)
    launch_gemm(stream, simA, simBT, comb,
                (long long)TT, (long long)TT, (long long)TT, 0, 0,
                TQ, TQ, TQ, TQ, TQ, TQ, BHQ,
                nullptr, 1.0f, nullptr, 0, 0, 0.0f, GM_F32);

    // softmax rows -> bf16 attention
    softmax_rows_kernel<<<dim3(BHQ * TQ), dim3(256), 0, stream>>>(comb, att, TQ);

    // attn @ V (Bt = v^T), fused head-merge into [B,T,A] bf16
    launch_gemm(stream, att, vT, attm,
                (long long)TT, sQK,
                (long long)TQ * AQ, (long long)HDQ, 4,
                TQ, TQ, AQ, TQ, HDQ, TQ, BHQ,
                nullptr, 1.0f, nullptr, 0, 0, 0.0f, GM_BF16);

    // output projection: attno = attm @ Wo + bo  (Bt = Wo^T)
    launch_gemm(stream, attm, WoT, attno, 0, 0, 0, 0, 0,
                AQ, AQ, DQ, BTQ, DQ, AQ, 1,
                bo, 1.0f, nullptr, 0, 0, 0.0f, GM_F32);

    // o1 = LN(x + attn)
    add_ln_kernel<<<dim3(BTQ), dim3(256), 0, stream>>>(attno, x_f, g1, be1,
                                                       o1f, o1b, DQ);

    // h = relu(o1 @ W1 + b1)  (Bt = W1^T, bf16 out)
    launch_gemm(stream, o1b, W1T, hb, 0, 0, 0, 0, 0,
                DQ, DQ, FQ, BTQ, FQ, DQ, 1,
                b1, 1.0f, nullptr, 0, 0, 0.0f, GM_BF16_RELU);

    // ffn = h @ W2 + b2  (Bt = W2^T)
    launch_gemm(stream, hb, W2T, ffn, 0, 0, 0, 0, 0,
                FQ, FQ, DQ, BTQ, DQ, FQ, 1,
                b2, 1.0f, nullptr, 0, 0, 0.0f, GM_F32);

    // out = LN(o1 + ffn)
    add_ln_kernel<<<dim3(BTQ), dim3(256), 0, stream>>>(ffn, o1f, g2, be2,
                                                       outs, nullptr, DQ);
  }
}
